// LocalCorrelation_3264175145315
// MI455X (gfx1250) — matrix operand order repaired and checked by reading, not compiled, not run on hardware
//
#include <hip/hip_runtime.h>
#include <hip/hip_bf16.h>

// ---------------------------------------------------------------------------
// Local correlation via CDNA5 WMMA + Tensor Data Mover (gfx1250, wave32).
//   corr[b, ky*13+kx, h, w] = <a_n[b,:,h,w], b_n[b,:,h+ky-6,w+kx-6]> / 0.07
// Neighborhood QK^T: per 16-pixel row tile and fixed ky,
//   S[16x32] = Q(16x256) * K(32x256)^T   (8 chunks of K=32 channels)
// then extract diagonals S[i][i+kx], kx=0..12.
// Keys for one (tile,ky) are one contiguous 16KB block of the padded BHWC
// buffer -> staged into LDS by TDM (tensor_load_to_lds), double-buffered so
// the DMA of ky+1 overlaps the WMMAs of ky.
// ---------------------------------------------------------------------------

typedef __attribute__((ext_vector_type(16))) _Float16 v16h;
typedef __attribute__((ext_vector_type(8)))  _Float16 v8h;
typedef __attribute__((ext_vector_type(4)))  _Float16 v4h;
typedef __attribute__((ext_vector_type(8)))  float    v8f;
typedef __attribute__((ext_vector_type(4)))  unsigned int u32x4;
typedef __attribute__((ext_vector_type(8)))  int          i32x8;
typedef __attribute__((ext_vector_type(4)))  int          i32x4;

#define RADIUS 6
#define KSZ    13      // 2*RADIUS+1
#define NB     4
#define NC     256
#define NH     128
#define NW     128
#define PH     (NH + 2 * RADIUS)        // 140
#define PW     (NW + 2 * RADIUS + 10)   // 150: left pad 6, right pad 16
#define INV_T  (1.0f / 0.07f)

#if __has_builtin(__builtin_amdgcn_tensor_load_to_lds) && \
    __has_builtin(__builtin_amdgcn_s_wait_tensorcnt)
#define USE_TDM 1
#else
#define USE_TDM 0
#endif

// ---------------------------------------------------------------------------
// Kernel A: per-pixel inverse L2 norm over channels (BCHW, coalesced along W)
// ---------------------------------------------------------------------------
__global__ __launch_bounds__(256) void norms_kernel(
    const float* __restrict__ fa, const float* __restrict__ fb,
    float* __restrict__ inva, float* __restrict__ invb) {
  int p = blockIdx.x * 256 + threadIdx.x;          // 0 .. B*H*W-1, w fastest
  if (p >= NB * NH * NW) return;
  int b  = p / (NH * NW);
  int hw = p - b * (NH * NW);
  size_t base = (size_t)b * NC * NH * NW + hw;
  float sa = 0.f, sb = 0.f;
  for (int c = 0; c < NC; ++c) {
    float va = fa[base + (size_t)c * (NH * NW)];
    float vb = fb[base + (size_t)c * (NH * NW)];
    sa += va * va;
    sb += vb * vb;
  }
  inva[p] = 1.0f / fmaxf(sqrtf(sa), 1e-12f);
  invb[p] = 1.0f / fmaxf(sqrtf(sb), 1e-12f);
}

// ---------------------------------------------------------------------------
// Zero-fill (for the padded key buffer), 16B vector grid-stride
// ---------------------------------------------------------------------------
__global__ void fill0(float4* __restrict__ p, int n) {
  for (int i = blockIdx.x * blockDim.x + threadIdx.x; i < n;
       i += gridDim.x * blockDim.x)
    p[i] = make_float4(0.f, 0.f, 0.f, 0.f);
}

// ---------------------------------------------------------------------------
// Kernel B: scale by inv-norm, convert f32->f16, transpose BCHW -> B,OH,OW,C
// ---------------------------------------------------------------------------
__global__ __launch_bounds__(256) void pack_kernel(
    const float* __restrict__ in, const float* __restrict__ inv,
    _Float16* __restrict__ outp, int OH, int OW, int oh0, int ow0) {
  __shared__ _Float16 tile[32][33];
  const int x  = threadIdx.x & 31;   // w lane
  const int y  = threadIdx.x >> 5;   // 0..7 channel sub-lane
  const int blk = blockIdx.x;        // over B*H*(W/32) = 2048
  const int w0 = (blk & 3) * 32;
  const int h  = (blk >> 2) & (NH - 1);
  const int b  = blk >> 9;

  const size_t pix   = (size_t)(b * NH + h) * NW + w0;
  const float  scale = inv[pix + x];
  const size_t ibase = (size_t)b * NC * NH * NW + (size_t)h * NW + w0 + x;

  const int wl = threadIdx.x >> 3;        // 0..31 output pixel
  const int c4 = (threadIdx.x & 7) * 4;   // 0,4,..,28

  for (int chunk = 0; chunk < NC; chunk += 32) {
#pragma unroll
    for (int cc = 0; cc < 4; ++cc) {
      int cl  = y + cc * 8;
      float v = in[ibase + (size_t)(chunk + cl) * (NH * NW)] * scale;
      tile[cl][x] = (_Float16)v;
    }
    __syncthreads();
    v4h o;
    o[0] = tile[c4 + 0][wl];
    o[1] = tile[c4 + 1][wl];
    o[2] = tile[c4 + 2][wl];
    o[3] = tile[c4 + 3][wl];
    _Float16* dst = outp +
        (((size_t)(b * OH + h + oh0) * OW) + (w0 + ow0 + wl)) * NC + chunk + c4;
    *(v4h*)dst = o;
    __syncthreads();
  }
}

// ---------------------------------------------------------------------------
// TDM: 1-tile descriptor copying one contiguous 16KB key row block to LDS.
//   data_size = 8B units; tile = 2048 x 1; tensor_dim0 == tile_dim0 (no OOB).
//   6-arg toolchain form: (g0 u32x4, g1 i32x8, g2 i32x4, g3 i32x4, i32x8, cpol)
// ---------------------------------------------------------------------------
#if USE_TDM
__device__ __forceinline__ void tdm_load_row16k(const _Float16* src,
                                                unsigned lds_off) {
  const unsigned long long ga = (unsigned long long)(uintptr_t)src;
  u32x4 g0;
  g0.x = 1u;                                   // count=1, user descriptor
  g0.y = lds_off;                              // LDS byte address
  g0.z = (unsigned)(ga & 0xFFFFFFFFu);         // global_addr[31:0]
  g0.w = (unsigned)((ga >> 32) & 0x1FFFFFFu)   // global_addr[56:32]
         | (2u << 30);                         // type = 2 ("image")
  i32x8 g1;
  g1[0] = (3 << 16);                           // data_size = 8B; no flags
  g1[1] = (int)(2048u << 16);                  // tensor_dim0[15:0] in [31:16]
  g1[2] = (int)(1u << 16);                     // tensor_dim0[31:16]=0 | tensor_dim1=1
  g1[3] = (int)(2048u << 16);                  // tensor_dim1[31:16]=0 | tile_dim0=2048
  g1[4] = 1;                                   // tile_dim1=1, tile_dim2=0
  g1[5] = 2048;                                // tensor_dim0_stride[31:0]
  g1[6] = 0;
  g1[7] = 0;
  const i32x4 z4 = {0, 0, 0, 0};
  const i32x8 z8 = {0, 0, 0, 0, 0, 0, 0, 0};
  __builtin_amdgcn_tensor_load_to_lds(g0, g1, z4, z4, z8, 0);
}
#endif

// ---------------------------------------------------------------------------
// Main kernel: one wave (32 threads) per (b, h, 16-wide w tile); loops ky.
//   A frag (16x32 f16): lane m=l&15 row m; lo lanes K 0-15, hi 16-31.
//   B frag (32x16 f16): lane = key pixel n=l&15 (+16 blk1), 16 consecutive
//     channels at chunk*32 + (l>>4)*16.
//   C/D (16x16 f32): VGPR r, lane l -> M=r+8*(l>>4), N=l&15.
// ---------------------------------------------------------------------------
__global__ __launch_bounds__(32) void corr_wmma(
    const _Float16* __restrict__ qn,   // [B][H][W][C]
    const _Float16* __restrict__ kn,   // [B][PH][PW][C], shifted by +RADIUS
    float* __restrict__ out) {         // [B][169][H][W]
#if USE_TDM
  __shared__ __align__(64) _Float16 kbuf[2][32 * NC];  // 2 x 16KB double buffer
#endif
  __shared__ float sred[16 * 32];

  const int lane = threadIdx.x;             // 0..31
  const int tile = blockIdx.x;              // 0..4095
  const int w0 = (tile & 7) * 16;
  const int h  = (tile >> 3) & (NH - 1);
  const int b  = tile >> 10;

  const int m   = lane & 15;
  const int sel = lane >> 4;

  // --- Q fragments, register resident across all 13 ky ---
  const _Float16* qpix = qn + (((size_t)(b * NH + h) * NW + w0 + m) * NC);
  v16h qf[8];
#pragma unroll
  for (int c = 0; c < 8; ++c) {
    const _Float16* p = qpix + c * 32 + sel * 16;  // contiguous A layout
    v8h lo = *(const v8h*)(p);
    v8h hi = *(const v8h*)(p + 8);
    qf[c] = __builtin_shufflevector(lo, hi, 0, 1, 2, 3, 4, 5, 6, 7, 8, 9, 10,
                                    11, 12, 13, 14, 15);
  }

  // padded coords: key row = h+ky ; key col start = w0 (RADIUS folded in)
  const _Float16* krow0 = kn + (((size_t)(b * PH + h) * PW + w0) * NC);

#if USE_TDM
  // Prime the pipeline: ky=0 into buffer 0.
  tdm_load_row16k(krow0, (unsigned)(uintptr_t)&kbuf[0][0]);
  __builtin_amdgcn_s_wait_tensorcnt(0);
  asm volatile("" ::: "memory");
#endif

  for (int ky = 0; ky < KSZ; ++ky) {
#if USE_TDM
    if (ky + 1 < KSZ)   // prefetch ky+1 into the other buffer (DMA || WMMA)
      tdm_load_row16k(krow0 + (size_t)(ky + 1) * PW * NC,
                      (unsigned)(uintptr_t)&kbuf[(ky + 1) & 1][0]);
    const _Float16* kb = &kbuf[ky & 1][0];
    const _Float16* kp0 = kb + (size_t)m * NC + sel * 16;
    const _Float16* kp1 = kb + (size_t)(m + 16) * NC + sel * 16;
#else
    const _Float16* krow = krow0 + (size_t)ky * PW * NC;
    const _Float16* kp0 = krow + (size_t)m * NC + sel * 16;
    const _Float16* kp1 = krow + (size_t)(m + 16) * NC + sel * 16;
#endif

    v8f acc0 = {};
    v8f acc1 = {};
#pragma unroll
    for (int c = 0; c < 8; ++c) {
      v16h b0 = *(const v16h*)(kp0 + c * 32);
      v16h b1 = *(const v16h*)(kp1 + c * 32);
      acc0 = __builtin_amdgcn_wmma_f32_16x16x32_f16(false, qf[c], false, b0,
                                                    (short)0, acc0, false, false);
      acc1 = __builtin_amdgcn_wmma_f32_16x16x32_f16(false, qf[c], false, b1,
                                                    (short)0, acc1, false, false);
    }

    // stage S[16x32] into LDS (wave-private; split DS counter, no barrier).
    // This s_wait_dscnt also drains all kbuf reads of this iteration before
    // the next iteration's TDM overwrite of that buffer can matter.
#pragma unroll
    for (int r = 0; r < 8; ++r) {
      const int M = r + sel * 8;
      sred[M * 32 + m]      = acc0[r];
      sred[M * 32 + 16 + m] = acc1[r];
    }
    asm volatile("s_wait_dscnt 0" ::: "memory");

    // diagonal extraction: corr[i][kx] = S[i][i+kx]; lanes split kx by parity
    for (int kx = sel; kx < KSZ; kx += 2) {
      float v = sred[m * 32 + m + kx] * INV_T;
      out[(((size_t)b * (KSZ * KSZ) + ky * KSZ + kx) * NH + h) * NW + w0 + m] = v;
    }
    asm volatile("" ::: "memory");

#if USE_TDM
    // next buffer must be resident before it is consumed
    __builtin_amdgcn_s_wait_tensorcnt(0);
    asm volatile("" ::: "memory");
#endif
  }
}

// ---------------------------------------------------------------------------
extern "C" void kernel_launch(void* const* d_in, const int* in_sizes, int n_in,
                              void* d_out, int out_size, void* d_ws,
                              size_t ws_size, hipStream_t stream) {
  const float* fa = (const float*)d_in[0];
  const float* fb = (const float*)d_in[1];
  float* out = (float*)d_out;

  // Carve workspace (256B aligned):
  //   qn:   B*H*W*C f16   = 33,554,432 B
  //   kn:   B*PH*PW*C f16 = 43,008,000 B
  //   inva/invb: B*H*W f32 = 262,144 B each
  char* ws = (char*)d_ws;
  size_t off = 0;
  auto carve = [&](size_t bytes) {
    void* p = ws + off;
    off = (off + bytes + 255) & ~(size_t)255;
    return p;
  };
  _Float16* qn   = (_Float16*)carve((size_t)NB * NH * NW * NC * 2);
  _Float16* kn   = (_Float16*)carve((size_t)NB * PH * PW * NC * 2);
  float*    inva = (float*)carve((size_t)NB * NH * NW * 4);
  float*    invb = (float*)carve((size_t)NB * NH * NW * 4);

  const int npix = NB * NH * NW;                       // 65536
  norms_kernel<<<(npix + 255) / 256, 256, 0, stream>>>(fa, fb, inva, invb);

  const int nvec4 = (int)(((size_t)NB * PH * PW * NC * 2) / 16);
  fill0<<<2048, 256, 0, stream>>>((float4*)kn, nvec4);

  pack_kernel<<<NB * NH * (NW / 32), 256, 0, stream>>>(fa, inva, qn, NH, NW, 0, 0);
  pack_kernel<<<NB * NH * (NW / 32), 256, 0, stream>>>(fb, invb, kn, PH, PW,
                                                       RADIUS, RADIUS);

  const int ntiles = NB * NH * (NW / 16);              // 4096 tiles
  corr_wmma<<<ntiles, 32, 0, stream>>>(qn, kn, out);
}
